// GINetRecon_rich_top_4183298146468
// MI455X (gfx1250) — compile-verified, matrix operand order checked by reading, not run on hardware
//
#include <hip/hip_runtime.h>
#include <hip/hip_bf16.h>
#include <math.h>

// ---------------------------------------------------------------------------
// GINE GNN forward for MI455X (gfx1250, wave32, WMMA).
// GEMMs run through V_WMMA_F32_16X16X32_BF16 (bf16 in, f32 accum) with
// LDS-staged tiles pre-swizzled into fragment layout so every fragment is
// two aligned ds_load_b128s. Gather/scatter streams with
// global_atomic_add_f32 (agg buffer is L2-resident: 64MB << 192MB L2).
// ---------------------------------------------------------------------------

typedef unsigned short u16;
typedef unsigned int   u32;

typedef __attribute__((ext_vector_type(16))) __bf16          v16bf;
typedef __attribute__((ext_vector_type(8)))  unsigned int    v8u;
typedef __attribute__((ext_vector_type(4)))  unsigned int    v4u;
typedef __attribute__((ext_vector_type(8)))  float           v8f;

#define NNODES 50000
#define NEDGES 800000
#define NBATCH 2000
#define EMB    300
#define EMBP   320      // padded (mult of 64)
#define HID    600
#define HIDP   640      // padded (mult of 64)
#define LAY    5
#define EPS_BN 1e-5f

#define ACT_NONE     0
#define ACT_RELU     1
#define ACT_SOFTPLUS 2

__device__ __forceinline__ u16 f2bf(float f) {
  u32 u = __float_as_uint(f);
  u32 r = u + 0x7FFFu + ((u >> 16) & 1u);   // round-to-nearest-even
  return (u16)(r >> 16);
}
__device__ __forceinline__ float bf2f(u16 h) {
  return __uint_as_float(((u32)h) << 16);
}
__device__ __forceinline__ float apply_act(float v, int act) {
  if (act == ACT_RELU)     return fmaxf(v, 0.f);
  if (act == ACT_SOFTPLUS) return (v > 20.f) ? v : log1pf(__expf(v));
  return v;
}

// ---------------------------------------------------------------------------
// WMMA GEMM: C[M,Nc] = act(A_bf16[M,K] * B_bf16[K,Nc] + bias)
// Block: 256 threads = 8 waves; 128x64 block tile; each wave owns a 32x32
// subtile (2x2 accumulators). K staged through LDS in 32-wide steps.
//   sA : row-major, row stride 48 u16 (96B, mult of 16) -> A fragment =
//        two aligned ds_load_b128 at byte offsets half*16 and 32+half*16.
//   sBp: fragment-swizzled [col][K-pair] dwords, col stride 20 u32 (80B) ->
//        B fragment = two aligned ds_load_b128 (pairs half*4.., 8+half*4..).
// Requires: Nc multiple of 64, K multiple of 32 (callers pad). M guarded.
// ---------------------------------------------------------------------------
__global__ __launch_bounds__(256)
void gemm_bf16_wmma(const u16* __restrict__ A, const u16* __restrict__ Bm,
                    const float* __restrict__ bias, int nbias,
                    float* __restrict__ Cf, u16* __restrict__ Cb,
                    int M, int Ncols, int K, int lda, int ldb, int ldc, int act)
{
  __shared__ __align__(16) u16 sA[128 * 48];   // 128 rows x 32 cols (stride 48)
  __shared__ __align__(16) u32 sBp[64 * 20];   // 64 cols x 16 K-pairs (stride 20)

  const int t    = threadIdx.x;
  const int wid  = t >> 5;
  const int lane = t & 31;
  const int half = lane >> 4;   // K-offset group select (ISA 16-bit layout)
  const int ln   = lane & 15;   // M (for A) / N (for B,C) index within tile
  const int wm   = wid & 3;     // wave M-tile (0..3) -> rows wm*32..+32
  const int wn   = wid >> 2;    // wave N-tile (0..1) -> cols wn*32..+32
  const int row0 = blockIdx.y * 128;
  const int col0 = blockIdx.x * 64;

  v8f acc00 = {}, acc01 = {}, acc10 = {}, acc11 = {};

  for (int k0 = 0; k0 < K; k0 += 32) {
    // prefetch next K-step tiles into L2 (global_prefetch_b8)
    if (k0 + 32 < K) {
      __builtin_prefetch(Bm + (size_t)(k0 + 32) * ldb + col0, 0, 0);
      int pr = row0 + (t >> 1);
      if (pr < M) __builtin_prefetch(A + (size_t)pr * lda + k0 + 32, 0, 0);
    }
    __syncthreads();
    // ---- stage A tile: 128x32 bf16 = 2048 dwords, 8 per thread, coalesced
#pragma unroll
    for (int i = 0; i < 8; ++i) {
      int u   = i * 256 + t;
      int row = u >> 4;
      int col = (u & 15) * 2;
      int gr  = row0 + row;
      u32 val = 0u;
      if (gr < M) val = *(const u32*)(A + (size_t)gr * lda + k0 + col);
      *(u32*)(sA + row * 48 + col) = val;
    }
    // ---- stage B tile 32x64 pre-swizzled into [col][K-pair] dwords
    // task = (pair p in 0..15, col-pair cp in 0..31); 2 tasks per thread
#pragma unroll
    for (int i = 0; i < 2; ++i) {
      int task = i * 256 + t;
      int p    = task >> 5;          // K pair index
      int cp   = task & 31;          // column pair
      int k    = k0 + 2 * p;
      int gc   = col0 + 2 * cp;
      u32 d0 = *(const u32*)(Bm + (size_t)k * ldb + gc);        // cols gc,gc+1 @ k
      u32 d1 = *(const u32*)(Bm + (size_t)(k + 1) * ldb + gc);  // cols gc,gc+1 @ k+1
      u32 lo = (d0 & 0xffffu) | (d1 << 16);                 // {B[k+1][gc],B[k][gc]}
      u32 hi = (d0 >> 16) | (d1 & 0xffff0000u);             // {B[k+1][gc+1],B[k][gc+1]}
      sBp[(2 * cp) * 20 + p]     = lo;
      sBp[(2 * cp + 1) * 20 + p] = hi;
    }
    __syncthreads();

    // ---- A fragments (two 16-row blocks), each two aligned b128 LDS loads
    v8u au0, au1;
    {
      const u16* r0 = sA + (wm * 32 + ln) * 48;
      const u16* r1 = sA + (wm * 32 + 16 + ln) * 48;
      v4u lo0 = *(const v4u*)(r0 + half * 8);
      v4u hi0 = *(const v4u*)(r0 + 16 + half * 8);
      v4u lo1 = *(const v4u*)(r1 + half * 8);
      v4u hi1 = *(const v4u*)(r1 + 16 + half * 8);
#pragma unroll
      for (int j = 0; j < 4; ++j) {
        au0[j] = lo0[j]; au0[4 + j] = hi0[j];
        au1[j] = lo1[j]; au1[4 + j] = hi1[j];
      }
    }
    // ---- B fragments (two 16-col blocks), each two aligned b128 LDS loads
    v8u bu0, bu1;
    {
      const u32* c0p = sBp + (wn * 32 + ln) * 20;
      const u32* c1p = sBp + (wn * 32 + 16 + ln) * 20;
      v4u lo0 = *(const v4u*)(c0p + half * 4);
      v4u hi0 = *(const v4u*)(c0p + 8 + half * 4);
      v4u lo1 = *(const v4u*)(c1p + half * 4);
      v4u hi1 = *(const v4u*)(c1p + 8 + half * 4);
#pragma unroll
      for (int j = 0; j < 4; ++j) {
        bu0[j] = lo0[j]; bu0[4 + j] = hi0[j];
        bu1[j] = lo1[j]; bu1[4 + j] = hi1[j];
      }
    }
    v16bf a0 = __builtin_bit_cast(v16bf, au0);
    v16bf a1 = __builtin_bit_cast(v16bf, au1);
    v16bf b0 = __builtin_bit_cast(v16bf, bu0);
    v16bf b1 = __builtin_bit_cast(v16bf, bu1);

    acc00 = __builtin_amdgcn_wmma_f32_16x16x32_bf16(false, a0, false, b0,
                                                    (short)0, acc00, false, false);
    acc01 = __builtin_amdgcn_wmma_f32_16x16x32_bf16(false, a0, false, b1,
                                                    (short)0, acc01, false, false);
    acc10 = __builtin_amdgcn_wmma_f32_16x16x32_bf16(false, a1, false, b0,
                                                    (short)0, acc10, false, false);
    acc11 = __builtin_amdgcn_wmma_f32_16x16x32_bf16(false, a1, false, b1,
                                                    (short)0, acc11, false, false);
  }

  // ---- epilogue: C vgpr r -> row = r + half*8 within 16x16 tile, col = ln
#pragma unroll
  for (int mi = 0; mi < 2; ++mi) {
#pragma unroll
    for (int r = 0; r < 8; ++r) {
      int row = row0 + wm * 32 + mi * 16 + half * 8 + r;
      if (row >= M) continue;
      int c0 = col0 + wn * 32 + ln;
      int c1 = c0 + 16;
      float v0 = (mi ? acc10[r] : acc00[r]) + ((c0 < nbias) ? bias[c0] : 0.f);
      float v1 = (mi ? acc11[r] : acc01[r]) + ((c1 < nbias) ? bias[c1] : 0.f);
      v0 = apply_act(v0, act);
      v1 = apply_act(v1, act);
      size_t o0 = (size_t)row * ldc + c0;
      size_t o1 = (size_t)row * ldc + c1;
      if (Cf) { Cf[o0] = v0;       Cf[o1] = v1; }
      if (Cb) { Cb[o0] = f2bf(v0); Cb[o1] = f2bf(v1); }
    }
  }
}

// ---------------------------------------------------------------------------
// Elementwise / helper kernels
// ---------------------------------------------------------------------------
__global__ void zero_f32(float* p, int n) {
  for (int i = blockIdx.x * blockDim.x + threadIdx.x; i < n;
       i += gridDim.x * blockDim.x) p[i] = 0.f;
}

__global__ void f32_to_bf16(const float* __restrict__ a, u16* __restrict__ b, int n) {
  for (int i = blockIdx.x * blockDim.x + threadIdx.x; i < n;
       i += gridDim.x * blockDim.x) b[i] = f2bf(a[i]);
}

// pad + convert weight matrix W[K,Nn] (row-major f32) -> Wb[Kp,Np] bf16
__global__ void wconv_pad(const float* __restrict__ W, u16* __restrict__ Wb,
                          int K, int Nn, int Kp, int Np) {
  int tot = Kp * Np;
  for (int i = blockIdx.x * blockDim.x + threadIdx.x; i < tot;
       i += gridDim.x * blockDim.x) {
    int k = i / Np, c = i - k * Np;
    float v = (k < K && c < Nn) ? W[(size_t)k * Nn + c] : 0.f;
    Wb[i] = f2bf(v);
  }
}

// h[n] = atom_e1[x0] + atom_e2[x1] + x[2:42] @ atom_w3 + b3  (padded cols 0)
__global__ __launch_bounds__(128)
void node_embed(const int* __restrict__ x, const float* __restrict__ e1,
                const float* __restrict__ e2, const float* __restrict__ w3,
                const float* __restrict__ b3, float* __restrict__ h) {
  __shared__ int xr[42];
  int n = blockIdx.x;
  if (threadIdx.x < 42) xr[threadIdx.x] = x[(size_t)n * 42 + threadIdx.x];
  __syncthreads();
  for (int c = threadIdx.x; c < EMBP; c += 128) {
    float v = 0.f;
    if (c < EMB) {
      v = e1[xr[0] * EMB + c] + e2[xr[1] * EMB + c] + b3[c];
#pragma unroll
      for (int j = 0; j < 40; ++j) v += (float)xr[2 + j] * w3[j * EMB + c];
    }
    h[(size_t)n * EMBP + c] = v;
  }
}

// one block per edge (incl. implicit self loops e >= NEDGES):
// agg[dst] += h[src] + edge_e1[a0] + edge_e2[a1] + eaf@w3 + b3
__global__ __launch_bounds__(128)
void edge_scatter(const float* __restrict__ h, const int* __restrict__ eidx,
                  const int* __restrict__ eattr,
                  const float* __restrict__ e1, const float* __restrict__ e2,
                  const float* __restrict__ w3, const float* __restrict__ b3,
                  float* __restrict__ agg) {
  __shared__ int   s_src, s_dst, s_a0, s_a1;
  __shared__ float s_eaf[8];
  int e = blockIdx.x;
  if (threadIdx.x == 0) {
    if (e < NEDGES) {
      s_src = eidx[e];
      s_dst = eidx[NEDGES + e];
      s_a0  = eattr[(size_t)e * 10 + 0];
      s_a1  = eattr[(size_t)e * 10 + 1];
    } else {               // self loop: attr col0 = 4, rest 0
      int i = e - NEDGES;
      s_src = i; s_dst = i; s_a0 = 4; s_a1 = 0;
    }
  }
  if (threadIdx.x < 8)
    s_eaf[threadIdx.x] =
        (e < NEDGES) ? (float)eattr[(size_t)e * 10 + 2 + threadIdx.x] : 0.f;
  __syncthreads();
  const float* hs = h + (size_t)s_src * EMBP;
  float* ag = agg + (size_t)s_dst * EMBP;
  for (int c = threadIdx.x; c < EMB; c += 128) {
    float v = hs[c] + e1[s_a0 * EMB + c] + e2[s_a1 * EMB + c] + b3[c];
#pragma unroll
    for (int j = 0; j < 8; ++j) v += s_eaf[j] * w3[j * EMB + c];
    atomicAdd(ag + c, v);          // global_atomic_add_f32 (L2-resident)
  }
}

// BN column stats: one block per channel; buffer is L2-resident so the
// column-strided reads amortize across the 300 channel blocks.
__global__ __launch_bounds__(256)
void bn_stats(const float* __restrict__ hraw, const float* __restrict__ g,
              const float* __restrict__ b, float* __restrict__ stats) {
  __shared__ float ss[256], ss2[256];
  int c = blockIdx.x;     // 0..EMB-1
  float s = 0.f, s2 = 0.f;
  for (int r = threadIdx.x; r < NNODES; r += 256) {
    float v = hraw[(size_t)r * EMBP + c];
    s += v; s2 += v * v;
  }
  ss[threadIdx.x] = s; ss2[threadIdx.x] = s2;
  __syncthreads();
  for (int o = 128; o > 0; o >>= 1) {
    if (threadIdx.x < o) { ss[threadIdx.x] += ss[threadIdx.x + o];
                           ss2[threadIdx.x] += ss2[threadIdx.x + o]; }
    __syncthreads();
  }
  if (threadIdx.x == 0) {
    float mu  = ss[0] / (float)NNODES;
    float var = ss2[0] / (float)NNODES - mu * mu;
    float sc  = g[c] * rsqrtf(var + EPS_BN);
    stats[c]        = sc;
    stats[EMBP + c] = b[c] - mu * sc;
  }
}

__global__ void bn_apply(const float* __restrict__ hraw,
                         const float* __restrict__ stats,
                         float* __restrict__ h, int do_relu) {
  int tot = NNODES * EMBP;
  for (int i = blockIdx.x * blockDim.x + threadIdx.x; i < tot;
       i += gridDim.x * blockDim.x) {
    int c = i % EMBP;
    float v = 0.f;
    if (c < EMB) {
      v = hraw[i] * stats[c] + stats[EMBP + c];
      if (do_relu) v = fmaxf(v, 0.f);
    }
    h[i] = v;
  }
}

__global__ void copy_h_out(const float* __restrict__ h, float* __restrict__ out) {
  int tot = NNODES * EMB;
  for (int i = blockIdx.x * blockDim.x + threadIdx.x; i < tot;
       i += gridDim.x * blockDim.x) {
    int n = i / EMB, c = i - n * EMB;
    out[i] = h[(size_t)n * EMBP + c];
  }
}

__global__ void pool_count(const int* __restrict__ batch, float* __restrict__ cnt) {
  for (int n = blockIdx.x * blockDim.x + threadIdx.x; n < NNODES;
       n += gridDim.x * blockDim.x) atomicAdd(&cnt[batch[n]], 1.f);
}

__global__ void pool_sum(const float* __restrict__ h, const int* __restrict__ batch,
                         float* __restrict__ pool) {
  int tot = NNODES * EMB;
  for (int i = blockIdx.x * blockDim.x + threadIdx.x; i < tot;
       i += gridDim.x * blockDim.x) {
    int n = i / EMB, c = i - n * EMB;
    atomicAdd(&pool[(size_t)batch[n] * EMBP + c], h[(size_t)n * EMBP + c]);
  }
}

// xcat[b, 0:512] = [pooled(300) | x_add(200) | zero pad(12)] as bf16
__global__ void build_xcat(const float* __restrict__ pool,
                           const float* __restrict__ cnt,
                           const float* __restrict__ x_add,
                           u16* __restrict__ xcat) {
  int tot = NBATCH * 512;
  for (int i = blockIdx.x * blockDim.x + threadIdx.x; i < tot;
       i += gridDim.x * blockDim.x) {
    int b = i >> 9, c = i & 511;
    float v = 0.f;
    if (c < EMB)            v = pool[(size_t)b * EMBP + c] / fmaxf(cnt[b], 1.f);
    else if (c < EMB + 200) v = x_add[(size_t)b * 200 + (c - EMB)];
    xcat[i] = f2bf(v);
  }
}

__global__ __launch_bounds__(64)
void pred_final(const u16* __restrict__ g2, const float* __restrict__ pw3,
                const float* __restrict__ pb3, float* __restrict__ outp) {
  __shared__ float red[64];
  int b = blockIdx.x;
  float s = 0.f;
  for (int c = threadIdx.x; c < 256; c += 64)
    s += bf2f(g2[(size_t)b * 256 + c]) * pw3[c];
  red[threadIdx.x] = s;
  __syncthreads();
  for (int o = 32; o > 0; o >>= 1) {
    if (threadIdx.x < o) red[threadIdx.x] += red[threadIdx.x + o];
    __syncthreads();
  }
  if (threadIdx.x == 0) outp[b] = red[0] + pb3[0];
}

// ---------------------------------------------------------------------------
static inline void launch_gemm(const u16* A, const u16* Bm, const float* bias,
                               int nbias, float* Cf, u16* Cb, int M, int Nc,
                               int K, int lda, int ldb, int ldc, int act,
                               hipStream_t s) {
  dim3 g((Nc + 63) / 64, (M + 127) / 128);
  gemm_bf16_wmma<<<g, 256, 0, s>>>(A, Bm, bias, nbias, Cf, Cb,
                                   M, Nc, K, lda, ldb, ldc, act);
}

extern "C" void kernel_launch(void* const* d_in, const int* in_sizes, int n_in,
                              void* d_out, int out_size, void* d_ws, size_t ws_size,
                              hipStream_t stream) {
  (void)in_sizes; (void)n_in; (void)out_size; (void)ws_size;

  const int*   x       = (const int*)d_in[0];
  const int*   eidx    = (const int*)d_in[1];
  const int*   eattr   = (const int*)d_in[2];
  const int*   batch   = (const int*)d_in[3];
  const float* x_add   = (const float*)d_in[4];
  const float* atom_e1 = (const float*)d_in[5];
  const float* atom_e2 = (const float*)d_in[6];
  const float* atom_w3 = (const float*)d_in[7];
  const float* atom_b3 = (const float*)d_in[8];
  const float* edge_e1 = (const float*)d_in[9];    // [L,6,EMB]
  const float* edge_e2 = (const float*)d_in[10];   // [L,4,EMB]
  const float* edge_w3 = (const float*)d_in[11];   // [L,8,EMB]
  const float* edge_b3 = (const float*)d_in[12];   // [L,EMB]
  const float* mlp_w1  = (const float*)d_in[13];   // [L,EMB,HID]
  const float* mlp_b1  = (const float*)d_in[14];   // [L,HID]
  const float* mlp_w2  = (const float*)d_in[15];   // [L,HID,EMB]
  const float* mlp_b2  = (const float*)d_in[16];   // [L,EMB]
  const float* bn_g    = (const float*)d_in[17];   // [L,EMB]
  const float* bn_b    = (const float*)d_in[18];   // [L,EMB]
  const float* feat_w  = (const float*)d_in[19];   // [500,512]
  const float* feat_b  = (const float*)d_in[20];   // [512]
  const float* pw1     = (const float*)d_in[21];   // [512,256]
  const float* pb1     = (const float*)d_in[22];
  const float* pw2     = (const float*)d_in[23];   // [256,256]
  const float* pb2     = (const float*)d_in[24];
  const float* pw3     = (const float*)d_in[25];   // [256,1]
  const float* pb3     = (const float*)d_in[26];
  float* out = (float*)d_out;                      // [N*EMB | B]

  // ---- workspace carve-out (~234 MB) ----
  char* w = (char*)d_ws;
  float* h     = (float*)w; w += (size_t)NNODES * EMBP * 4;   // node features (padded)
  float* buf2  = (float*)w; w += (size_t)NNODES * EMBP * 4;   // agg / pre-BN
  u16*   abf   = (u16*)w;   w += (size_t)NNODES * EMBP * 2;   // bf16 agg
  u16*   hidb  = (u16*)w;   w += (size_t)NNODES * HIDP * 2;   // bf16 hidden
  u16*   wb1   = (u16*)w;   w += (size_t)512 * 512 * 2;       // bf16 weight A
  u16*   wb2   = (u16*)w;   w += (size_t)HIDP * EMBP * 2;     // bf16 weight B
  float* stats = (float*)w; w += (size_t)2 * EMBP * 4;        // BN scale/shift
  float* pool  = (float*)w; w += (size_t)NBATCH * EMBP * 4;
  float* cnt   = (float*)w; w += (size_t)NBATCH * 4;
  u16*   xcat  = (u16*)w;   w += (size_t)NBATCH * 512 * 2;
  u16*   g0b   = (u16*)w;   w += (size_t)NBATCH * 512 * 2;
  u16*   g1b   = (u16*)w;   w += (size_t)NBATCH * 256 * 2;
  u16*   g2b   = (u16*)w;   w += (size_t)NBATCH * 256 * 2;

  const int GS = 2048;   // grid for grid-stride kernels

  // ---- node embedding ----
  node_embed<<<NNODES, 128, 0, stream>>>(x, atom_e1, atom_e2, atom_w3, atom_b3, h);

  // ---- GINE layers ----
  for (int l = 0; l < LAY; ++l) {
    zero_f32<<<GS, 256, 0, stream>>>(buf2, NNODES * EMBP);
    edge_scatter<<<NEDGES + NNODES, 128, 0, stream>>>(
        h, eidx, eattr,
        edge_e1 + (size_t)l * 6 * EMB, edge_e2 + (size_t)l * 4 * EMB,
        edge_w3 + (size_t)l * 8 * EMB, edge_b3 + (size_t)l * EMB, buf2);

    f32_to_bf16<<<GS, 256, 0, stream>>>(buf2, abf, NNODES * EMBP);
    wconv_pad<<<GS, 256, 0, stream>>>(mlp_w1 + (size_t)l * EMB * HID, wb1,
                                      EMB, HID, EMBP, HIDP);
    wconv_pad<<<GS, 256, 0, stream>>>(mlp_w2 + (size_t)l * HID * EMB, wb2,
                                      HID, EMB, HIDP, EMBP);

    // MLP: relu(agg @ w1 + b1) @ w2 + b2   (both via WMMA)
    launch_gemm(abf, wb1, mlp_b1 + (size_t)l * HID, HID, nullptr, hidb,
                NNODES, HIDP, EMBP, EMBP, HIDP, HIDP, ACT_RELU, stream);
    launch_gemm(hidb, wb2, mlp_b2 + (size_t)l * EMB, EMB, buf2, nullptr,
                NNODES, EMBP, HIDP, HIDP, EMBP, EMBP, ACT_NONE, stream);

    // BatchNorm (batch stats) + ReLU except last layer
    bn_stats<<<EMB, 256, 0, stream>>>(buf2, bn_g + (size_t)l * EMB,
                                      bn_b + (size_t)l * EMB, stats);
    bn_apply<<<GS, 256, 0, stream>>>(buf2, stats, h, (l != LAY - 1) ? 1 : 0);
  }

  // ---- emit h_node ----
  copy_h_out<<<GS, 256, 0, stream>>>(h, out);

  // ---- global mean pool ----
  zero_f32<<<GS, 256, 0, stream>>>(pool, NBATCH * EMBP);
  zero_f32<<<64, 256, 0, stream>>>(cnt, NBATCH);
  pool_count<<<GS, 256, 0, stream>>>(batch, cnt);
  pool_sum<<<GS, 256, 0, stream>>>(h, batch, pool);
  build_xcat<<<GS, 256, 0, stream>>>(pool, cnt, x_add, xcat);

  // ---- head: feat_lin + softplus MLP (all WMMA except final 256->1 dot) ----
  wconv_pad<<<GS, 256, 0, stream>>>(feat_w, wb1, 500, 512, 512, 512);
  launch_gemm(xcat, wb1, feat_b, 512, nullptr, g0b,
              NBATCH, 512, 512, 512, 512, 512, ACT_NONE, stream);

  wconv_pad<<<GS, 256, 0, stream>>>(pw1, wb2, 512, 256, 512, 256);
  launch_gemm(g0b, wb2, pb1, 256, nullptr, g1b,
              NBATCH, 256, 512, 512, 256, 256, ACT_SOFTPLUS, stream);

  wconv_pad<<<GS, 256, 0, stream>>>(pw2, wb2, 256, 256, 256, 256);
  launch_gemm(g1b, wb2, pb2, 256, nullptr, g2b,
              NBATCH, 256, 256, 256, 256, 256, ACT_SOFTPLUS, stream);

  pred_final<<<NBATCH, 64, 0, stream>>>(g2b, pw3, pb3,
                                        out + (size_t)NNODES * EMB);
}
